// up_sample_39934605918324
// MI455X (gfx1250) — matrix operand order repaired and checked by reading, not compiled, not run on hardware
//
#include <hip/hip_runtime.h>
#include <hip/hip_bf16.h>

// Fused Conv1d(64,64,3)+ReLU -> Conv1d(64,4,3)+ReLU -> phase-shift(r=4)
// for x:[32,64,16384] fp32.  bf16 WMMA (f32 accum) with 3-term hi/lo split
// for fp32-grade accuracy; everything between the global read of x and the
// global write of the interleaved output lives in LDS.

typedef __attribute__((ext_vector_type(16))) __bf16 v16bf;
typedef __attribute__((ext_vector_type(8)))  float  v8f;

#define LLEN   16384
#define CCH    64
#define BATCHN 32
#define TL     128          // output positions per workgroup
#define XS_P   176          // staged x positions: [l0-24, l0+152)
#define H1_P   160          // staged h1 positions: [l0-16, l0+144)
#define PITCH  72           // channel pitch (ushorts): 144B rows, 16B aligned
#define NT1    10           // GEMM1 n-tiles (160 positions)
#define KT     6            // K tiles: 192 / 32

__device__ __forceinline__ unsigned short bf16_rne(float f) {
  unsigned u = __float_as_uint(f);
  unsigned r = u + 0x7fffu + ((u >> 16) & 1u);   // round-to-nearest-even
  return (unsigned short)(r >> 16);
}
__device__ __forceinline__ float bfbits(unsigned short h) {
  return __uint_as_float(((unsigned)h) << 16);
}

union FragAB { v16bf v; uint4 q[2]; };
union FragC  { v8f  v; float f[8]; };

struct alignas(16) Smem {
  unsigned short xsHi[XS_P * PITCH];   // x tile, bf16 high part  [pos][ci]
  unsigned short xsLo[XS_P * PITCH];   // x tile, bf16 low  part
  unsigned short h1Hi[H1_P * PITCH];   // relu(conv1) bf16 high   [pos][co]
  unsigned short h1Lo[H1_P * PITCH];
  unsigned short wa1Hi[24 * 512];      // w1 A-fragments (4 mt x 6 kt)
  unsigned short wa1Lo[24 * 512];
  unsigned short wa2Hi[6 * 512];       // w2 A-fragments (1 mt x 6 kt)
  unsigned short wa2Lo[6 * 512];
  float b1s[64];
  float b2s[4];
};

__global__ __launch_bounds__(256)
void fused_upsample_kernel(const float* __restrict__ x,
                           const float* __restrict__ w1,
                           const float* __restrict__ b1,
                           const float* __restrict__ w2,
                           const float* __restrict__ b2,
                           float* __restrict__ out)
{
  __shared__ Smem sm;

  const int tid   = threadIdx.x;
  const int bid   = blockIdx.x;
  const int batch = bid >> 7;            // 128 tiles per batch
  const int l0    = (bid & 127) * TL;
  const size_t xbase = (size_t)batch * CCH * LLEN;

  // ---------------- stage biases ----------------
  if (tid < 64) sm.b1s[tid] = b1[tid];
  if (tid < 4)  sm.b2s[tid] = b2[tid];

  // ---------------- stage x tile (fp32 -> bf16 hi/lo), coalesced in pos ----
  for (int i = tid; i < XS_P * CCH; i += 256) {
    int ci = i / XS_P, p = i - ci * XS_P;
    int gpos = l0 - 24 + p;
    float v = (gpos >= 0 && gpos < LLEN) ? x[xbase + (size_t)ci * LLEN + gpos]
                                         : 0.0f;
    unsigned short h = bf16_rne(v);
    sm.xsHi[p * PITCH + ci] = h;
    sm.xsLo[p * PITCH + ci] = bf16_rne(v - bfbits(h));
  }

  // ---------------- stage w1 A-fragments (contiguous k order: K = 16*half + e)
  // logical reduction index q = k*64 + ci  (so B loads are channel-contiguous)
  for (int i = tid; i < 24 * 512; i += 256) {
    int frag = i >> 9;                    // mt*6 + kt
    int lane = (i >> 4) & 31;
    int e    = i & 15;
    int mt = frag / 6, kt = frag - mt * 6;
    int K = e + ((lane >= 16) ? 16 : 0);
    int q = kt * 32 + K;
    int k = q >> 6, ci = q & 63;
    int co = mt * 16 + (lane & 15);
    float v = w1[((size_t)co * CCH + ci) * 3 + k];
    unsigned short h = bf16_rne(v);
    sm.wa1Hi[i] = h;
    sm.wa1Lo[i] = bf16_rne(v - bfbits(h));
  }
  // ---------------- stage w2 A-fragments (rows co>=4 are zero)
  for (int i = tid; i < 6 * 512; i += 256) {
    int kt   = i >> 9;
    int lane = (i >> 4) & 31;
    int e    = i & 15;
    int K = e + ((lane >= 16) ? 16 : 0);
    int q = kt * 32 + K;
    int k = q >> 6, ci = q & 63;
    int co = lane & 15;
    float v = (co < 4) ? w2[((size_t)co * CCH + ci) * 3 + k] : 0.0f;
    unsigned short h = bf16_rne(v);
    sm.wa2Hi[i] = h;
    sm.wa2Lo[i] = bf16_rne(v - bfbits(h));
  }
  __syncthreads();

  const int wave   = tid >> 5;
  const int lane   = tid & 31;
  const int lanelo = lane & 15;
  const int laneHi = lane >> 4;

  // ---------------- GEMM1: h1 = relu(W1 * im2col(x) + b1), N span 160 ------
  for (int item = wave; item < 4 * NT1; item += 8) {
    int mt = item / NT1, nt = item - mt * NT1;
    FragC c;
#pragma unroll
    for (int r = 0; r < 8; r++) c.f[r] = sm.b1s[mt * 16 + laneHi * 8 + r];

#pragma unroll
    for (int kt = 0; kt < KT; kt++) {
      FragAB ah, al, bh, bl;
      const uint4* pa  = (const uint4*)&sm.wa1Hi[((mt * 6 + kt) * 32 + lane) * 16];
      const uint4* pal = (const uint4*)&sm.wa1Lo[((mt * 6 + kt) * 32 + lane) * 16];
      ah.q[0] = pa[0];  ah.q[1] = pa[1];
      al.q[0] = pal[0]; al.q[1] = pal[1];

      int k  = kt >> 1;
      int cb = (kt & 1) * 32 + laneHi * 16;
      int pl = 8 + nt * 16 + lanelo + k - 1;      // xs local position
      const uint4* pb  = (const uint4*)&sm.xsHi[pl * PITCH + cb];
      const uint4* pbl = (const uint4*)&sm.xsLo[pl * PITCH + cb];
      bh.q[0] = pb[0];  bh.q[1] = pb[1];
      bl.q[0] = pbl[0]; bl.q[1] = pbl[1];

      c.v = __builtin_amdgcn_wmma_f32_16x16x32_bf16(false, ah.v, false, bh.v,
                                                    (short)0, c.v, false, false);
      c.v = __builtin_amdgcn_wmma_f32_16x16x32_bf16(false, ah.v, false, bl.v,
                                                    (short)0, c.v, false, false);
      c.v = __builtin_amdgcn_wmma_f32_16x16x32_bf16(false, al.v, false, bh.v,
                                                    (short)0, c.v, false, false);
    }

    // relu + split to bf16 hi/lo; zero positions outside [0,L) (conv padding)
    int  gpos  = l0 - 16 + nt * 16 + lanelo;
    bool valid = (gpos >= 0) && (gpos < LLEN);
    unsigned hw[4], lw[4];
#pragma unroll
    for (int r2 = 0; r2 < 4; r2++) {
      float f0 = valid ? fmaxf(c.f[2 * r2], 0.f)     : 0.f;
      float f1 = valid ? fmaxf(c.f[2 * r2 + 1], 0.f) : 0.f;
      unsigned short h0 = bf16_rne(f0), h1 = bf16_rne(f1);
      hw[r2] = (unsigned)h0 | ((unsigned)h1 << 16);
      unsigned short q0 = bf16_rne(f0 - bfbits(h0));
      unsigned short q1 = bf16_rne(f1 - bfbits(h1));
      lw[r2] = (unsigned)q0 | ((unsigned)q1 << 16);
    }
    int hoff = (nt * 16 + lanelo) * PITCH + mt * 16 + laneHi * 8;
    *(uint4*)&sm.h1Hi[hoff] = make_uint4(hw[0], hw[1], hw[2], hw[3]);
    *(uint4*)&sm.h1Lo[hoff] = make_uint4(lw[0], lw[1], lw[2], lw[3]);
  }
  __syncthreads();

  // ---------------- GEMM2: out = phase_shift(relu(W2 * im2col(h1) + b2)) ---
  {
    int nt = wave;                                  // 8 n-tiles, 1 per wave
    FragC c;
#pragma unroll
    for (int r = 0; r < 8; r++) {
      int co2 = laneHi * 8 + r;
      c.f[r] = (co2 < 4) ? sm.b2s[co2] : 0.f;
    }
#pragma unroll
    for (int kt = 0; kt < KT; kt++) {
      FragAB ah, al, bh, bl;
      const uint4* pa  = (const uint4*)&sm.wa2Hi[(kt * 32 + lane) * 16];
      const uint4* pal = (const uint4*)&sm.wa2Lo[(kt * 32 + lane) * 16];
      ah.q[0] = pa[0];  ah.q[1] = pa[1];
      al.q[0] = pal[0]; al.q[1] = pal[1];

      int k  = kt >> 1;
      int cb = (kt & 1) * 32 + laneHi * 16;
      int pl = 16 + nt * 16 + lanelo + k - 1;       // h1s local position
      const uint4* pb  = (const uint4*)&sm.h1Hi[pl * PITCH + cb];
      const uint4* pbl = (const uint4*)&sm.h1Lo[pl * PITCH + cb];
      bh.q[0] = pb[0];  bh.q[1] = pb[1];
      bl.q[0] = pbl[0]; bl.q[1] = pbl[1];

      c.v = __builtin_amdgcn_wmma_f32_16x16x32_bf16(false, ah.v, false, bh.v,
                                                    (short)0, c.v, false, false);
      c.v = __builtin_amdgcn_wmma_f32_16x16x32_bf16(false, ah.v, false, bl.v,
                                                    (short)0, c.v, false, false);
      c.v = __builtin_amdgcn_wmma_f32_16x16x32_bf16(false, al.v, false, bh.v,
                                                    (short)0, c.v, false, false);
    }
    // Phase shift r=4: out[b][pos*4 + co2] -> lane's 4 channels are one float4
    if (laneHi == 0) {
      int pos = l0 + nt * 16 + lanelo;
      float4 o;
      o.x = fmaxf(c.f[0], 0.f);
      o.y = fmaxf(c.f[1], 0.f);
      o.z = fmaxf(c.f[2], 0.f);
      o.w = fmaxf(c.f[3], 0.f);
      ((float4*)out)[(size_t)batch * LLEN + pos] = o;
    }
  }
}

extern "C" void kernel_launch(void* const* d_in, const int* in_sizes, int n_in,
                              void* d_out, int out_size, void* d_ws, size_t ws_size,
                              hipStream_t stream) {
  (void)in_sizes; (void)n_in; (void)d_ws; (void)ws_size; (void)out_size;
  const float* x  = (const float*)d_in[0];
  const float* w1 = (const float*)d_in[1];
  const float* b1 = (const float*)d_in[2];
  const float* w2 = (const float*)d_in[3];
  const float* b2 = (const float*)d_in[4];
  // d_in[5] = sc (fixed at 4 in setup_inputs; layout above assumes r=4)
  float* out = (float*)d_out;

  dim3 grid(BATCHN * (LLEN / TL));   // 4096 workgroups
  fused_upsample_kernel<<<grid, 256, 0, stream>>>(x, w1, b1, w2, b2, out);
}
